// GCNShared_69801808495209
// MI455X (gfx1250) — matrix operand order repaired and checked by reading, not compiled, not run on hardware
//
#include <hip/hip_runtime.h>
#include <hip/hip_bf16.h>

typedef __bf16 bf16;
typedef __attribute__((ext_vector_type(16))) __bf16 v16bf;
typedef __attribute__((ext_vector_type(8)))  __bf16 v8bf;
typedef __attribute__((ext_vector_type(8)))  float  v8f;

#define BS    16
#define L     128
#define D     256
#define LDS_W 264   // padded LDS row stride in bf16 elems (multiple of 8 -> 16B aligned rows)
#define NSTEP 3

// ---------------------------------------------------------------------------
// Prepack: fp32 weights -> bf16, sliced for the three live GEMMs.
//   wbf[0]       = agg_w[e][k]           (256x256)
//   wbf[1*D*D]   = W1  = attn_w[:, :D]
//   wbf[2*D*D]   = U1  = upd_w[:, :D]
//   wbf[3*D*D]   = U2  = upd_w[:, D:]
// (attn_b and W2 cancel out of the softmax and are never needed.)
// ---------------------------------------------------------------------------
__global__ void prepack_weights(const float* __restrict__ agg_w,
                                const float* __restrict__ attn_w,
                                const float* __restrict__ upd_w,
                                bf16* __restrict__ wbf) {
  int idx = blockIdx.x * blockDim.x + threadIdx.x;
  if (idx >= D * D) return;
  int e = idx >> 8;
  int k = idx & (D - 1);
  wbf[idx]           = (bf16)agg_w[idx];
  wbf[D*D   + idx]   = (bf16)attn_w[e * (2 * D) + k];
  wbf[2*D*D + idx]   = (bf16)upd_w[e * (2 * D) + k];
  wbf[3*D*D + idx]   = (bf16)upd_w[e * (2 * D) + D + k];
}

// ---------------------------------------------------------------------------
// WMMA fragment loaders (16x16x32 bf16, wave32; this code's contiguous k order, same for A and B).
// A (16x32, MxK): lane l -> row l&15 ; elems 0..15 = K[kb..kb+15] contiguous
//                 with kb = k0 + 16*(l>>4).
// B (32x16, KxN): lane l -> col l&15 ; elems 0..15 = K[k0+16*(l>>4) .. +15] contiguous.
// C/D (16x16 f32): elem r -> row 16*mt + r + 8*(l>>4), col 16*nt + (l&15).
// ---------------------------------------------------------------------------
__device__ __forceinline__ v16bf load_a_frag(const bf16* A, int mtile, int k0, int lane) {
  int row = mtile * 16 + (lane & 15);
  int kb  = k0 + ((lane >> 4) << 4);
  const bf16* p = A + row * LDS_W + kb;
  v8bf lo = *(const v8bf*)(p);
  v8bf hi = *(const v8bf*)(p + 8);
  v16bf r;
#pragma unroll
  for (int i = 0; i < 8; ++i) { r[i] = lo[i]; r[i + 8] = hi[i]; }
  return r;
}

__device__ __forceinline__ v16bf load_b_frag(const bf16* W, int ntile, int k0, int lane) {
  int col = ntile * 16 + (lane & 15);
  int kb  = k0 + ((lane >> 4) << 4);
  return *(const v16bf*)(W + col * D + kb);   // 32B contiguous, 32B aligned
}

// 4 n-tiles at a time: one A fragment per K-step reused across 4 B fragments.
// Live set ~ 4*8 (acc) + 8 (A) + B in flight : bounded, no spills.
__device__ __forceinline__ void gemm4(const bf16* A, const bf16* W,
                                      int mtile, int nto, int lane, v8f acc[4]) {
#pragma unroll
  for (int k0 = 0; k0 < D; k0 += 32) {
    v16bf a = load_a_frag(A, mtile, k0, lane);
#pragma unroll
    for (int j = 0; j < 4; ++j) {
      v16bf b = load_b_frag(W, nto + j, k0, lane);
      acc[j] = __builtin_amdgcn_wmma_f32_16x16x32_bf16(false, a, false, b,
                                                       (short)0, acc[j], false, false);
    }
  }
}

// ---------------------------------------------------------------------------
// Fused 3-step GCN kernel: one workgroup per batch, hidden state in LDS.
// ---------------------------------------------------------------------------
__global__ __launch_bounds__(256, 1)
void gcn_fused(const float* __restrict__ feat, const float* __restrict__ mask,
               const float* __restrict__ agg_b, const float* __restrict__ upd_b,
               const bf16* __restrict__ wbf, float* __restrict__ out) {
  // bufA is triple-purposed across barrier-separated phases:
  //   hid (masked)  ->  s1  ->  hid_next
  __shared__ bf16  bufA[L * LDS_W];
  __shared__ bf16  xhid[L * LDS_W];
  __shared__ float maskL[L];
  __shared__ bf16  aggL[D];
  __shared__ float vcol[D];

  const int b    = blockIdx.x;
  const int tid  = threadIdx.x;
  const int lane = tid & 31;
  const int wid  = tid >> 5;          // wave id == M-tile (8 waves x 16 rows = 128)

  const bf16* Wagg = wbf;
  const bf16* W1   = wbf + D * D;
  const bf16* U1   = wbf + 2 * D * D;
  const bf16* U2   = wbf + 3 * D * D;

  if (tid < L) maskL[tid] = mask[b * L + tid];

  // feat * mask -> bufA (bf16)   [mask of step 0 folded in]
  const float* fb = feat + (size_t)b * L * D;
  for (int idx = tid; idx < L * D; idx += 256) {
    int i = idx >> 8, dd = idx & (D - 1);
    bufA[i * LDS_W + dd] = (bf16)(fb[idx] * mask[b * L + i]);
  }
  __syncthreads();

  for (int step = 0; step < NSTEP; ++step) {
    const bool last = (step == NSTEP - 1);
    const int  colb  = lane & 15;
    const int  rbase = wid * 16 + ((lane >> 4) << 3);

    // ---- phase A: xhid = hid @ agg_w^T + agg_b -------------------------
#pragma clang loop unroll(disable)
    for (int nto = 0; nto < 16; nto += 4) {
      v8f acc[4];
#pragma unroll
      for (int j = 0; j < 4; ++j) {
        float bias = agg_b[(nto + j) * 16 + colb];
#pragma unroll
        for (int r = 0; r < 8; ++r) acc[j][r] = bias;
      }
      gemm4(bufA, Wagg, wid, nto, lane, acc);
#pragma unroll
      for (int j = 0; j < 4; ++j) {
        int col = (nto + j) * 16 + colb;
#pragma unroll
        for (int r = 0; r < 8; ++r)
          xhid[(rbase + r) * LDS_W + col] = (bf16)acc[j][r];
      }
    }
    __syncthreads();

    // ---- phase B: s1 = xhid @ W1^T  -> bufA (hid now dead) -------------
#pragma clang loop unroll(disable)
    for (int nto = 0; nto < 16; nto += 4) {
      v8f acc[4];
#pragma unroll
      for (int j = 0; j < 4; ++j)
#pragma unroll
        for (int r = 0; r < 8; ++r) acc[j][r] = 0.0f;
      gemm4(xhid, W1, wid, nto, lane, acc);
#pragma unroll
      for (int j = 0; j < 4; ++j) {
        int col = (nto + j) * 16 + colb;
#pragma unroll
        for (int r = 0; r < 8; ++r)
          bufA[(rbase + r) * LDS_W + col] = (bf16)acc[j][r];
      }
    }
    __syncthreads();

    // ---- phase C: per-column softmax over i, agg = sigmoid(sum p*xhid) -
    // (s2 and attn_b are constant along the softmax axis and cancel.)
    {
      int e = tid;                     // 256 threads <-> 256 feature columns
      float m = -3.4e38f;
      for (int i = 0; i < L; ++i)
        m = fmaxf(m, (float)bufA[i * LDS_W + e]);
      float denom = 0.0f, numer = 0.0f;
      for (int i = 0; i < L; ++i) {
        float p = __expf((float)bufA[i * LDS_W + e] - m);
        denom += p;
        numer += p * (float)xhid[i * LDS_W + e];
      }
      float a = numer / denom;
      aggL[e] = (bf16)(1.0f / (1.0f + __expf(-a)));
    }
    __syncthreads();

    // ---- phase D: vcol[e] = upd_b[e] + sum_d agg[d] * U2[e][d] ---------
    {
      int e = tid;
      const bf16* u2row = U2 + e * D;
      float acc = upd_b[e];
      for (int dd = 0; dd < D; ++dd)
        acc += (float)aggL[dd] * (float)u2row[dd];
      vcol[e] = acc;
    }
    __syncthreads();

    // ---- phase E: hid_next = xhid @ U1^T + vcol ------------------------
#pragma clang loop unroll(disable)
    for (int nto = 0; nto < 16; nto += 4) {
      v8f acc[4];
#pragma unroll
      for (int j = 0; j < 4; ++j) {
        float bias = vcol[(nto + j) * 16 + colb];
#pragma unroll
        for (int r = 0; r < 8; ++r) acc[j][r] = bias;
      }
      gemm4(xhid, U1, wid, nto, lane, acc);
      if (last) {
        float* ob = out + (size_t)b * L * D;
#pragma unroll
        for (int j = 0; j < 4; ++j) {
          int col = (nto + j) * 16 + colb;
#pragma unroll
          for (int r = 0; r < 8; ++r) ob[(rbase + r) * D + col] = acc[j][r];
        }
      } else {
        // fold next step's mask multiply into the store
#pragma unroll
        for (int j = 0; j < 4; ++j) {
          int col = (nto + j) * 16 + colb;
#pragma unroll
          for (int r = 0; r < 8; ++r)
            bufA[(rbase + r) * LDS_W + col] = (bf16)(acc[j][r] * maskL[rbase + r]);
        }
      }
    }
    __syncthreads();
  }
}

extern "C" void kernel_launch(void* const* d_in, const int* in_sizes, int n_in,
                              void* d_out, int out_size, void* d_ws, size_t ws_size,
                              hipStream_t stream) {
  (void)in_sizes; (void)n_in; (void)out_size; (void)ws_size;
  const float* feat   = (const float*)d_in[0];
  const float* mask   = (const float*)d_in[1];
  const float* agg_w  = (const float*)d_in[2];
  const float* agg_b  = (const float*)d_in[3];
  const float* attn_w = (const float*)d_in[4];
  // d_in[5] (attn_b) provably cancels in the softmax; unused.
  const float* upd_w  = (const float*)d_in[6];
  const float* upd_b  = (const float*)d_in[7];

  bf16* wbf = (bf16*)d_ws;   // 4 * 256*256 bf16 = 512 KB scratch

  prepack_weights<<<(D * D + 255) / 256, 256, 0, stream>>>(agg_w, attn_w, upd_w, wbf);
  gcn_fused<<<BS, 256, 0, stream>>>(feat, mask, agg_b, upd_b, wbf, (float*)d_out);
}
